// DPL_14207751815815
// MI455X (gfx1250) — matrix operand order repaired and checked by reading, not compiled, not run on hardware
//
#include <hip/hip_runtime.h>
#include <cstdint>

// MI455X / gfx1250, wave32. Layer-2 GEMM on v_wmma_f32_16x16x32_f16.
typedef __attribute__((ext_vector_type(16))) _Float16 v16h;
typedef __attribute__((ext_vector_type(8)))  float    v8f;

#define BATCH 1048576
#define HDIM  256
#define W2S   264              // padded f16 row stride: 132 dwords -> 4-bank lane skew, conflict-free

// Dynamic-LDS layout (bytes)
#define OFF_W2  0              // 256*264*2 = 135168  (w2 as f16, padded rows)
#define OFF_W1  135168         // 768*4    = 3072
#define OFF_B1  138240         // 256*4    = 1024
#define OFF_B2  139264         // 256*4    = 1024
#define OFF_W3  140288         // 768*4    = 3072
#define OFF_B3  143360         // 16
#define SMEM_BYTES 143376

__global__ __launch_bounds__(512)
void mlp_parity_wmma(const float* __restrict__ x,  const float* __restrict__ w1,
                     const float* __restrict__ b1, const float* __restrict__ w2,
                     const float* __restrict__ b2, const float* __restrict__ w3,
                     const float* __restrict__ b3, float* __restrict__ out)
{
    extern __shared__ char smem[];
    _Float16* sW2 = (_Float16*)(smem + OFF_W2);
    float*    sW1 = (float*)(smem + OFF_W1);
    float*    sB1 = (float*)(smem + OFF_B1);
    float*    sB2 = (float*)(smem + OFF_B2);
    float*    sW3 = (float*)(smem + OFF_W3);
    float*    sB3 = (float*)(smem + OFF_B3);

    const int tid = threadIdx.x;
    const int nthreads = blockDim.x;

    // ---- Stage weights into LDS (w2 converted f32->f16, padded rows) ----
    for (int i = tid; i < HDIM * HDIM; i += nthreads) {
        int n = i >> 8, k = i & 255;
        sW2[n * W2S + k] = (_Float16)w2[i];
    }
    for (int i = tid; i < HDIM * 3; i += nthreads) sW1[i] = w1[i];
    for (int i = tid; i < HDIM;     i += nthreads) { sB1[i] = b1[i]; sB2[i] = b2[i]; }
    for (int i = tid; i < 3 * HDIM; i += nthreads) sW3[i] = w3[i];
    if (tid < 3) sB3[tid] = b3[tid];
    __syncthreads();

    const int lane = tid & 31;
    const int half = lane >> 4;   // 0: lanes 0-15, 1: lanes 16-31
    const int nl   = lane & 15;
    const int wave   = (blockIdx.x * nthreads + tid) >> 5;
    const int nwaves = (gridDim.x * nthreads) >> 5;
    const int ntiles = BATCH / 16;

    // Per-lane base inside sW2 for B fragments: row = column index n, offset half*8 halves
    // (B 32x16 layout: lane half h holds K {8h+0..7, 16+8h+0..7}; N = lane&15)
    for (int tile = wave; tile < ntiles; tile += nwaves) {
        const long r0 = (long)tile * 16;

        // This lane's A-matrix row (M = lane&15 for both halves per ISA layout)
        const float* xr = x + (r0 + nl) * 3;
        const float x0 = xr[0], x1 = xr[1], x2 = xr[2];

        // ---- Layer 1 (K=3) computed directly into WMMA A-fragment layout ----
        // 16-bit A 16x32: lane<16 holds K {8h+0..7, 16+8h+0..7} of row M=lane&15.
        v16h a[8];
        #pragma unroll
        for (int kc = 0; kc < 8; ++kc) {
            union { v16h v; _Float16 h[16]; } af;
            #pragma unroll
            for (int j = 0; j < 16; ++j) {
                const int k = kc * 32 + ((j < 8) ? (8 * half + j)
                                                 : (16 + 8 * half + (j - 8)));
                float s = fmaf(x0, sW1[k * 3 + 0],
                          fmaf(x1, sW1[k * 3 + 1],
                          fmaf(x2, sW1[k * 3 + 2], sB1[k])));
                af.h[j] = (_Float16)fmaxf(s, 0.0f);
            }
            a[kc] = af.v;
        }

        // Layer-3 partial accumulators: rows r=0..7 (global row = 8*half+r), c=0..2
        float l3[8][3];
        #pragma unroll
        for (int r = 0; r < 8; ++r) { l3[r][0] = 0.f; l3[r][1] = 0.f; l3[r][2] = 0.f; }

        // ---- Layer 2: 16 N-tiles x 8 K-chunks of v_wmma_f32_16x16x32_f16 ----
        // KEEP ROLLED: full unroll spilled B fragments to scratch (VGPR cap at
        // 512 thr/block). Rolled: ~140 live VGPRs, B goes LDS -> WMMA directly.
        #pragma unroll 1
        for (int nt = 0; nt < 16; ++nt) {
            const int ng = nt * 16 + nl;           // this lane's output column N
            // Epilogue operands issued early so they overlap the WMMA chain
            const float b2v = sB2[ng];
            const float w30 = sW3[0 * HDIM + ng];
            const float w31 = sW3[1 * HDIM + ng];
            const float w32 = sW3[2 * HDIM + ng];
            const _Float16* bbase = sW2 + (size_t)ng * W2S + half * 8;

            v8f acc = {};
            #pragma unroll
            for (int kc = 0; kc < 8; ++kc) {
                // B[k][n] = w2[n][k]  (layer2 is h @ w2^T); two 16B runs per lane
                const _Float16* bp = bbase + kc * 32;
                union { v16h v; uint4 q[2]; } bf;
                bf.q[0] = *(const uint4*)(bp);
                bf.q[1] = *(const uint4*)(bp + 16);
                acc = __builtin_amdgcn_wmma_f32_16x16x32_f16(
                          false, a[kc], false, bf.v, (short)0, acc, false, false);
            }
            // Epilogue: bias + ReLU, fold into layer-3 (N=3) partials
            #pragma unroll
            for (int r = 0; r < 8; ++r) {
                const float h2 = fmaxf(acc[r] + b2v, 0.0f);
                l3[r][0] = fmaf(h2, w30, l3[r][0]);
                l3[r][1] = fmaf(h2, w31, l3[r][1]);
                l3[r][2] = fmaf(h2, w32, l3[r][2]);
            }
        }

        // ---- Reduce the 16 column-lanes within each half-group (masks < 16) ----
        #pragma unroll
        for (int r = 0; r < 8; ++r)
            #pragma unroll
            for (int c = 0; c < 3; ++c) {
                float v = l3[r][c];
                v += __shfl_xor(v, 1, 32);
                v += __shfl_xor(v, 2, 32);
                v += __shfl_xor(v, 4, 32);
                v += __shfl_xor(v, 8, 32);
                l3[r][c] = v;
            }

        // ---- Parity head: one lane per row writes logitC (3) + pred (2) ----
        const float b30 = sB3[0], b31 = sB3[1], b32 = sB3[2];
        if (nl < 8) {
            #pragma unroll
            for (int r = 0; r < 8; ++r) {
                if (nl == r) {
                    const long row = r0 + half * 8 + r;
                    const float L0 = l3[r][0] + b30;
                    const float L1 = l3[r][1] + b31;
                    const float L2 = l3[r][2] + b32;
                    // logitC at offset 2*BATCH (outputs concatenated: pred, logitC)
                    float* lc = out + 2L * BATCH + row * 3;
                    lc[0] = L0; lc[1] = L1; lc[2] = L2;

                    const float s0 = 1.0f / (1.0f + __expf(-L0));
                    const float s1 = 1.0f / (1.0f + __expf(-L1));
                    const float s2 = 1.0f / (1.0f + __expf(-L2));
                    // q_c = p_neg - p_pos = (1 - 2*sig)/(1 + 2*eps); P(odd) = (1 - q0 q1 q2)/2
                    const float inv_eps = 1.0f / (1.0f + 2.0f * 1e-5f);
                    const float q0 = (1.0f - 2.0f * s0) * inv_eps;
                    const float q1 = (1.0f - 2.0f * s1) * inv_eps;
                    const float q2 = (1.0f - 2.0f * s2) * inv_eps;
                    const float pred1 = 0.5f * (1.0f - q0 * q1 * q2);
                    const float invn = 1.0f / 1.002f;
                    out[row * 2 + 0] = (1.0f - pred1 + 0.001f) * invn;
                    out[row * 2 + 1] = (pred1 + 0.001f) * invn;
                }
            }
        }
    }
}

extern "C" void kernel_launch(void* const* d_in, const int* in_sizes, int n_in,
                              void* d_out, int out_size, void* d_ws, size_t ws_size,
                              hipStream_t stream) {
    (void)in_sizes; (void)n_in; (void)out_size; (void)d_ws; (void)ws_size;
    const float* x  = (const float*)d_in[0];
    const float* w1 = (const float*)d_in[1];
    const float* b1 = (const float*)d_in[2];
    const float* w2 = (const float*)d_in[3];
    const float* b2 = (const float*)d_in[4];
    const float* w3 = (const float*)d_in[5];
    const float* b3 = (const float*)d_in[6];
    float* out = (float*)d_out;

    // 512 blocks x 512 threads = 8192 waves; 65536 tiles -> 8 tiles/wave.
    // Amortizes the per-block w2->LDS staging; ~2 blocks/WGP by LDS (140KB/320KB).
    dim3 grid(512), block(512);
    hipLaunchKernelGGL(mlp_parity_wmma, grid, block, SMEM_BYTES, stream,
                       x, w1, b1, w2, b2, w3, b3, out);
}
